// ETLayer_37641093382253
// MI455X (gfx1250) — matrix operand order repaired and checked by reading, not compiled, not run on hardware
//
#include <hip/hip_runtime.h>

#define B_ 2
#define N_ 256
#define C_ 128
#define H_ 8
#define D_ 16
#define ROWS (B_*N_*N_)     // 131072 flattened (b,i,j) rows
#define NM (N_*N_)          // 65536
#define SLICES 17           // 16 d-channels + 1 denominator slice

typedef __bf16 bf16;
typedef __attribute__((ext_vector_type(16))) __bf16 v16bf;
typedef __attribute__((ext_vector_type(8)))  float  v8f;
typedef int gint4 __attribute__((vector_size(16)));   // raw clang vector, matches builtin param

#if defined(__has_builtin)
#  if __has_builtin(__builtin_amdgcn_global_load_async_to_lds_b128)
#    define USE_ASYNC_LDS 1
#  endif
#  if __has_builtin(__builtin_amdgcn_s_wait_asynccnt)
#    define HAVE_WAIT_ASYNC 1
#  endif
#endif

#if defined(USE_ASYNC_LDS)
static __device__ inline void async_copy16(const bf16* g, bf16* l) {
  // toolchain signature: (gint4* gsrc, gint4* ldst, imm offset, imm cpol)
  __builtin_amdgcn_global_load_async_to_lds_b128((gint4*)g, (gint4*)l, 0, 0);
}
static __device__ inline void async_wait0() {
#  if defined(HAVE_WAIT_ASYNC)
  __builtin_amdgcn_s_wait_asynccnt(0);
#  else
  asm volatile("s_wait_asynccnt 0x0" ::: "memory");
#  endif
}
#endif

// ---------------- WMMA fragment helpers (CDNA5 16x16x32 bf16 layouts) ----------------
// A (16x32, MxK): lane<16 -> row=lane, elems 0..15 = K[k0..k0+15]
//                 lane>=16 -> same row, K offset +16.
static __device__ inline v16bf fragA_f32(const float* p0, int ld, int lane, int k0) {
  int row = lane & 15;
  int ko  = k0 + ((lane & 16) ? 16 : 0);
  const float* p = p0 + (size_t)row * ld + ko;
  v16bf f;
#pragma unroll
  for (int e = 0; e < 8; e++) { f[e] = (bf16)p[e]; f[8 + e] = (bf16)p[8 + e]; }
  return f;
}
// B (32x16, KxN) from K-major [n][k] storage: lane%16 = n, lane>=16 -> K offset +16.
static __device__ inline v16bf fragB_f32(const float* p0, int ld, int lane, int k0) {
  int n  = lane & 15;
  int ko = k0 + ((lane & 16) ? 16 : 0);
  const float* p = p0 + (size_t)n * ld + ko;
  v16bf f;
#pragma unroll
  for (int e = 0; e < 16; e++) f[e] = (bf16)p[e];
  return f;
}
static __device__ inline v16bf fragA_bf(const bf16* p0, int ld, int lane, int k0) {
  int row = lane & 15;
  int ko  = k0 + ((lane & 16) ? 16 : 0);
  const bf16* p = p0 + (size_t)row * ld + ko;
  v16bf f;
  ((uint4*)&f)[0] = *(const uint4*)p;         // 8 bf16 = 16B
  ((uint4*)&f)[1] = *(const uint4*)(p + 8);  // next K chunk
  return f;
}
static __device__ inline v16bf fragB_bf(const bf16* p0, int ld, int lane, int k0) {
  int n  = lane & 15;
  int ko = k0 + ((lane & 16) ? 16 : 0);
  const bf16* p = p0 + (size_t)n * ld + ko;
  v16bf f;
  ((uint4*)&f)[0] = *(const uint4*)p;
  ((uint4*)&f)[1] = *(const uint4*)(p + 8);
  return f;
}
static __device__ inline v8f wmma_bf16(v16bf a, v16bf b, v8f c) {
  return __builtin_amdgcn_wmma_f32_16x16x32_bf16(false, a, false, b, (short)0, c,
                                                 false, false);
}

// ---------------- K1: projections a1,a2 (f32) and v1,v2 (bf16) ----------------
// one wave per 16 flattened rows; 4 waves/block
__global__ void k1_proj(const float* __restrict__ h,
                        const float* __restrict__ wa1, const float* __restrict__ wa2,
                        const float* __restrict__ wv1, const float* __restrict__ wv2,
                        float* __restrict__ A1, float* __restrict__ A2,
                        bf16* __restrict__ V1R, bf16* __restrict__ V2R) {
  int lane = threadIdx.x & 31;
  int wave = threadIdx.x >> 5;
  size_t row0 = ((size_t)blockIdx.x * 4 + wave) * 16;
  const float* hrow = h + row0 * C_;

  v16bf af[4];
#pragma unroll
  for (int kk = 0; kk < 4; kk++) af[kk] = fragA_f32(hrow, C_, lane, kk * 32);

  int n = lane & 15;
  int hi8 = (lane & 16) ? 8 : 0;

#pragma unroll
  for (int nt = 0; nt < 8; nt++) {  // v1 = h @ wv1^T
    v8f acc = {};
#pragma unroll
    for (int kk = 0; kk < 4; kk++)
      acc = wmma_bf16(af[kk], fragB_f32(wv1 + nt * 16 * C_, C_, lane, kk * 32), acc);
#pragma unroll
    for (int r = 0; r < 8; r++)
      V1R[(row0 + r + hi8) * C_ + nt * 16 + n] = (bf16)acc[r];
  }
#pragma unroll
  for (int nt = 0; nt < 8; nt++) {  // v2 = h @ wv2^T
    v8f acc = {};
#pragma unroll
    for (int kk = 0; kk < 4; kk++)
      acc = wmma_bf16(af[kk], fragB_f32(wv2 + nt * 16 * C_, C_, lane, kk * 32), acc);
#pragma unroll
    for (int r = 0; r < 8; r++)
      V2R[(row0 + r + hi8) * C_ + nt * 16 + n] = (bf16)acc[r];
  }
  {  // a: wa1 in cols 0..7, wa2 in cols 8..15 of one B tile
    v8f acc = {};
#pragma unroll
    for (int kk = 0; kk < 4; kk++) {
      const float* wr = (n < 8) ? (wa1 + n * C_) : (wa2 + (n - 8) * C_);
      int ko = kk * 32 + ((lane & 16) ? 16 : 0);
      v16bf bfrag;
#pragma unroll
      for (int e = 0; e < 16; e++) bfrag[e] = (bf16)wr[ko + e];
      acc = wmma_bf16(af[kk], bfrag, acc);
    }
#pragma unroll
    for (int r = 0; r < 8; r++) {
      size_t row = row0 + r + hi8;
      if (n < 8) A1[row * H_ + n] = acc[r];
      else       A2[row * H_ + (n - 8)] = acc[r];
    }
  }
}

// ---------------- K2: a1 row-max (over j) and a2 col-max (over i) ----------------
__global__ void k2_max(const float* __restrict__ A1, const float* __restrict__ A2,
                       float* __restrict__ M1, float* __restrict__ M2) {
  int idx = blockIdx.x * blockDim.x + threadIdx.x;  // 0 .. 2*B*N*H-1 = 8191
  int which = idx >> 12;
  int rem = idx & 4095;
  int b = rem >> 11;
  int r = (rem >> 3) & 255;
  int hh = rem & 7;
  float m = -INFINITY;
  if (which == 0) {
    const float* p = A1 + (((size_t)b * N_ + r) * N_) * H_ + hh;
    for (int j = 0; j < N_; j++) m = fmaxf(m, p[(size_t)j * H_]);
    M1[((size_t)b * N_ + r) * H_ + hh] = m;
  } else {
    const float* p = A2 + (size_t)b * NM * H_ + (size_t)r * H_ + hh;
    for (int i = 0; i < N_; i++) m = fmaxf(m, p[(size_t)i * N_ * H_]);
    M2[((size_t)b * N_ + r) * H_ + hh] = m;
  }
}

// ---------------- K3: mask+exp, scale v, repack K-major per (b,h,d) ----------------
// P[(bh*17+d)][i][l] = v1[b,i,l,h,d]*e1 ; P[..+16][i][l] = e1
// Q[(bh*17+d)][j][l] = v2[b,l,j,h,d]*e2 ; Q[..+16][j][l] = e2
__global__ void k3_pack(const float* __restrict__ A1, const float* __restrict__ A2,
                        const float* __restrict__ M1, const float* __restrict__ M2,
                        const unsigned char* __restrict__ mask,
                        const bf16* __restrict__ V1R, const bf16* __restrict__ V2R,
                        bf16* __restrict__ P, bf16* __restrict__ Q) {
  __shared__ __align__(16) bf16 tile[16][16][16];  // [d][x][y]
  int blk = blockIdx.x;
  int bh = blk >> 8;
  int trem = blk & 255;
  int x0 = (trem >> 4) << 4;
  int y0 = (trem & 15) << 4;
  int b = bh >> 3, hh = bh & 7;
  int t = threadIdx.x;
  int x = t >> 4, y = t & 15;

  {  // phase 1: P / E1  (x = i-offset, y = l-offset)
    int i = x0 + x, l = y0 + y;
    size_t rij = ((size_t)b * N_ + i) * N_ + l;
    float e1 = mask[rij] ? 0.f
                         : __expf(A1[rij * H_ + hh] - M1[((size_t)b * N_ + i) * H_ + hh]);
    P[((size_t)bh * SLICES + 16) * NM + (size_t)i * N_ + l] = (bf16)e1;
    const bf16* vp = V1R + rij * C_ + hh * D_;
#pragma unroll
    for (int dd = 0; dd < 16; dd++) tile[dd][x][y] = (bf16)((float)vp[dd] * e1);
  }
  __syncthreads();
  {
    int dd = t >> 4, xx = t & 15;
    const uint4* s = (const uint4*)&tile[dd][xx][0];
    uint4* g = (uint4*)(P + ((size_t)bh * SLICES + dd) * NM + (size_t)(x0 + xx) * N_ + y0);
    g[0] = s[0]; g[1] = s[1];
  }
  __syncthreads();
  {  // phase 2: Q / E2  (x = j-offset, y = l-offset)
    int j = x0 + x, l = y0 + y;
    size_t rlj = ((size_t)b * N_ + l) * N_ + j;
    float e2 = mask[rlj] ? 0.f
                         : __expf(A2[rlj * H_ + hh] - M2[((size_t)b * N_ + j) * H_ + hh]);
    Q[((size_t)bh * SLICES + 16) * NM + (size_t)j * N_ + l] = (bf16)e2;
    const bf16* vp = V2R + rlj * C_ + hh * D_;
#pragma unroll
    for (int dd = 0; dd < 16; dd++) tile[dd][x][y] = (bf16)((float)vp[dd] * e2);
  }
  __syncthreads();
  {
    int dd = t >> 4, xx = t & 15;
    const uint4* s = (const uint4*)&tile[dd][xx][0];
    uint4* g = (uint4*)(Q + ((size_t)bh * SLICES + dd) * NM + (size_t)(x0 + xx) * N_ + y0);
    g[0] = s[0]; g[1] = s[1];
  }
}

// ---------------- K4: triangle contraction, 272 batched 256x256x256 bf16 matmuls ----
// OUT[bhd][i][j] = sum_l P[bhd][i][l] * Q[bhd][j][l]
// 64x64 output tile per block; A/B slabs staged in LDS (64 KB) via async-to-LDS.
__global__ void k4_tri(const bf16* __restrict__ P, const bf16* __restrict__ Q,
                       float* __restrict__ OUT) {
  __shared__ __align__(16) bf16 Asl[64][256];
  __shared__ __align__(16) bf16 Bsl[64][256];
  int t = threadIdx.x;
  int lane = t & 31;
  int wave = t >> 5;
  int blk = blockIdx.x;
  int bhd = blk >> 4;
  int trem = blk & 15;
  int i0 = (trem >> 2) * 64;
  int j0 = (trem & 3) * 64;
  const bf16* Ag = P + (size_t)bhd * NM + (size_t)i0 * N_;
  const bf16* Bg = Q + (size_t)bhd * NM + (size_t)j0 * N_;

#pragma unroll
  for (int c = 0; c < 8; c++) {      // 2048 chunks of 8 bf16 per slab
    int idx = c * 256 + t;
    int row = idx >> 5;
    int col = (idx & 31) * 8;
#if defined(USE_ASYNC_LDS)
    async_copy16(Ag + (size_t)row * N_ + col, &Asl[row][col]);
    async_copy16(Bg + (size_t)row * N_ + col, &Bsl[row][col]);
#else
    *(uint4*)&Asl[row][col] = *(const uint4*)(Ag + (size_t)row * N_ + col);
    *(uint4*)&Bsl[row][col] = *(const uint4*)(Bg + (size_t)row * N_ + col);
#endif
  }
#if defined(USE_ASYNC_LDS)
  async_wait0();
#endif
  __syncthreads();

#pragma unroll
  for (int st = 0; st < 2; st++) {   // 8 waves x 2 subtiles = 4x4 grid of 16x16 tiles
    int tid2 = wave * 2 + st;
    int wi = (tid2 >> 2) * 16;
    int wj = (tid2 & 3) * 16;
    v8f acc = {};
#pragma unroll
    for (int k0 = 0; k0 < N_; k0 += 32)
      acc = wmma_bf16(fragA_bf(&Asl[wi][0], 256, lane, k0),
                      fragB_bf(&Bsl[wj][0], 256, lane, k0), acc);
    int n = lane & 15, hi8 = (lane & 16) ? 8 : 0;
    float* op = OUT + (size_t)bhd * NM + (size_t)(i0 + wi + hi8) * N_ + j0 + wj + n;
#pragma unroll
    for (int r = 0; r < 8; r++) op[(size_t)r * N_] = acc[r];
  }
}

// ---------------- K5: gather + /denom, wo GEMM, residual + LN1 ----------------
__global__ void k5_wo_ln1(const float* __restrict__ OUT, const float* __restrict__ h_in,
                          const float* __restrict__ wo,
                          const float* __restrict__ ln1s, const float* __restrict__ ln1b,
                          float* __restrict__ H1F, bf16* __restrict__ H1B) {
  __shared__ __align__(16) bf16 xb[16][C_];
  __shared__ __align__(16) float yf[16][C_];
  __shared__ float red[16][8];
  __shared__ float mu[16], rv[16];

  int blk = blockIdx.x;
  int b = blk >> 12;
  int rem = blk & 4095;
  int i = rem >> 4;
  int j0 = (rem & 15) << 4;
  size_t row0 = ((size_t)b * N_ + i) * N_ + j0;

  int t = threadIdx.x;  // 0..127
  int lane = t & 31, wave = t >> 5;

  {  // x[jj][c] = OUT/(denom+1e-6), c = h*16+d
    int hh = t >> 4, d = t & 15;
    const float* op = OUT + (((size_t)b * H_ + hh) * SLICES + d) * NM + (size_t)i * N_ + j0;
    const float* dp = OUT + (((size_t)b * H_ + hh) * SLICES + 16) * NM + (size_t)i * N_ + j0;
#pragma unroll
    for (int jj = 0; jj < 16; jj++)
      xb[jj][t] = (bf16)(op[jj] / (dp[jj] + 1e-6f));
  }
  __syncthreads();
#pragma unroll
  for (int s = 0; s < 2; s++) {  // y = x @ wo^T
    int nt = wave * 2 + s;
    v8f acc = {};
#pragma unroll
    for (int kk = 0; kk < 4; kk++)
      acc = wmma_bf16(fragA_bf(&xb[0][0], C_, lane, kk * 32),
                      fragB_f32(wo + nt * 16 * C_, C_, lane, kk * 32), acc);
    int n = lane & 15, hi8 = (lane & 16) ? 8 : 0;
#pragma unroll
    for (int r = 0; r < 8; r++) yf[r + hi8][nt * 16 + n] = acc[r];
  }
  __syncthreads();
  {  // residual
    for (int jj = 0; jj < 16; jj++) yf[jj][t] += h_in[(row0 + jj) * C_ + t];
  }
  __syncthreads();
  {  // LN mean
    int rr = t >> 3, sub = t & 7;
    float s0 = 0.f;
    for (int cc = sub * 16; cc < sub * 16 + 16; cc++) s0 += yf[rr][cc];
    red[rr][sub] = s0;
  }
  __syncthreads();
  if ((t & 7) == 0) {
    int rr = t >> 3;
    float s0 = 0.f;
    for (int k = 0; k < 8; k++) s0 += red[rr][k];
    mu[rr] = s0 * (1.f / C_);
  }
  __syncthreads();
  {  // LN var
    int rr = t >> 3, sub = t & 7;
    float m = mu[rr], s0 = 0.f;
    for (int cc = sub * 16; cc < sub * 16 + 16; cc++) { float dv = yf[rr][cc] - m; s0 += dv * dv; }
    red[rr][sub] = s0;
  }
  __syncthreads();
  if ((t & 7) == 0) {
    int rr = t >> 3;
    float s0 = 0.f;
    for (int k = 0; k < 8; k++) s0 += red[rr][k];
    rv[rr] = rsqrtf(s0 * (1.f / C_) + 1e-5f);
  }
  __syncthreads();
  {
    float sc = ln1s[t], bi_ = ln1b[t];
    for (int jj = 0; jj < 16; jj++) {
      float v = (yf[jj][t] - mu[jj]) * rv[jj] * sc + bi_;
      H1F[(row0 + jj) * C_ + t] = v;
      H1B[(row0 + jj) * C_ + t] = (bf16)v;
    }
  }
}

// ---------------- K6: MLP (GEMM+SiLU, GEMM), residual + LN2 ----------------
__global__ void k6_mlp(const float* __restrict__ H1F, const bf16* __restrict__ H1B,
                       const float* __restrict__ w1, const float* __restrict__ b1,
                       const float* __restrict__ w2, const float* __restrict__ b2,
                       const float* __restrict__ ln2s, const float* __restrict__ ln2b,
                       float* __restrict__ out) {
  __shared__ __align__(16) bf16 xb[16][C_];
  __shared__ __align__(16) bf16 ub[16][C_];
  __shared__ __align__(16) float yf[16][C_];
  __shared__ float red[16][8];
  __shared__ float mu[16], rv[16];

  size_t row0 = (size_t)blockIdx.x * 16;
  int t = threadIdx.x, lane = t & 31, wave = t >> 5;
  {  // stage h1 tile
    int rr = t >> 3, ch = (t & 7) * 16;
    *(uint4*)&xb[rr][ch]     = *(const uint4*)&H1B[(row0 + rr) * C_ + ch];
    *(uint4*)&xb[rr][ch + 8] = *(const uint4*)&H1B[(row0 + rr) * C_ + ch + 8];
  }
  __syncthreads();
#pragma unroll
  for (int s = 0; s < 2; s++) {  // up1 = silu(h1 @ w1^T + b1)
    int nt = wave * 2 + s;
    v8f acc = {};
#pragma unroll
    for (int kk = 0; kk < 4; kk++)
      acc = wmma_bf16(fragA_bf(&xb[0][0], C_, lane, kk * 32),
                      fragB_f32(w1 + nt * 16 * C_, C_, lane, kk * 32), acc);
    int n = lane & 15, hi8 = (lane & 16) ? 8 : 0;
    float bb = b1[nt * 16 + n];
#pragma unroll
    for (int r = 0; r < 8; r++) {
      float u = acc[r] + bb;
      ub[r + hi8][nt * 16 + n] = (bf16)(u / (1.f + __expf(-u)));
    }
  }
  __syncthreads();
#pragma unroll
  for (int s = 0; s < 2; s++) {  // up2 = up1 @ w2^T + b2
    int nt = wave * 2 + s;
    v8f acc = {};
#pragma unroll
    for (int kk = 0; kk < 4; kk++)
      acc = wmma_bf16(fragA_bf(&ub[0][0], C_, lane, kk * 32),
                      fragB_f32(w2 + nt * 16 * C_, C_, lane, kk * 32), acc);
    int n = lane & 15, hi8 = (lane & 16) ? 8 : 0;
    float bb = b2[nt * 16 + n];
#pragma unroll
    for (int r = 0; r < 8; r++) yf[r + hi8][nt * 16 + n] = acc[r] + bb;
  }
  __syncthreads();
  {
    for (int jj = 0; jj < 16; jj++) yf[jj][t] += H1F[(row0 + jj) * C_ + t];
  }
  __syncthreads();
  {
    int rr = t >> 3, sub = t & 7;
    float s0 = 0.f;
    for (int cc = sub * 16; cc < sub * 16 + 16; cc++) s0 += yf[rr][cc];
    red[rr][sub] = s0;
  }
  __syncthreads();
  if ((t & 7) == 0) {
    int rr = t >> 3;
    float s0 = 0.f;
    for (int k = 0; k < 8; k++) s0 += red[rr][k];
    mu[rr] = s0 * (1.f / C_);
  }
  __syncthreads();
  {
    int rr = t >> 3, sub = t & 7;
    float m = mu[rr], s0 = 0.f;
    for (int cc = sub * 16; cc < sub * 16 + 16; cc++) { float dv = yf[rr][cc] - m; s0 += dv * dv; }
    red[rr][sub] = s0;
  }
  __syncthreads();
  if ((t & 7) == 0) {
    int rr = t >> 3;
    float s0 = 0.f;
    for (int k = 0; k < 8; k++) s0 += red[rr][k];
    rv[rr] = rsqrtf(s0 * (1.f / C_) + 1e-5f);
  }
  __syncthreads();
  {
    float sc = ln2s[t], bi_ = ln2b[t];
    for (int jj = 0; jj < 16; jj++)
      out[(row0 + jj) * C_ + t] = (yf[jj][t] - mu[jj]) * rv[jj] * sc + bi_;
  }
}

// ---------------- host launcher ----------------
extern "C" void kernel_launch(void* const* d_in, const int* in_sizes, int n_in,
                              void* d_out, int out_size, void* d_ws, size_t ws_size,
                              hipStream_t stream) {
  (void)in_sizes; (void)n_in; (void)out_size; (void)ws_size;
  const float* h    = (const float*)d_in[0];
  const unsigned char* mask = (const unsigned char*)d_in[1];
  const float* wa1  = (const float*)d_in[2];
  const float* wa2  = (const float*)d_in[3];
  const float* wv1  = (const float*)d_in[4];
  const float* wv2  = (const float*)d_in[5];
  const float* wo   = (const float*)d_in[6];
  const float* ln1s = (const float*)d_in[7];
  const float* ln1b = (const float*)d_in[8];
  const float* ln2s = (const float*)d_in[9];
  const float* ln2b = (const float*)d_in[10];
  const float* w1   = (const float*)d_in[11];
  const float* b1   = (const float*)d_in[12];
  const float* w2   = (const float*)d_in[13];
  const float* b2   = (const float*)d_in[14];
  float* out = (float*)d_out;

  char* ws = (char*)d_ws;
  size_t off = 0;
  auto alloc = [&](size_t bytes) -> char* {
    char* p = ws + off;
    off = (off + bytes + 255) & ~(size_t)255;
    return p;
  };
  float* A1  = (float*)alloc((size_t)ROWS * H_ * sizeof(float));
  float* A2  = (float*)alloc((size_t)ROWS * H_ * sizeof(float));
  float* M1  = (float*)alloc((size_t)B_ * N_ * H_ * sizeof(float));
  float* M2  = (float*)alloc((size_t)B_ * N_ * H_ * sizeof(float));
  bf16*  V1R = (bf16*)alloc((size_t)ROWS * C_ * sizeof(bf16));
  bf16*  V2R = (bf16*)alloc((size_t)ROWS * C_ * sizeof(bf16));
  bf16*  P   = (bf16*)alloc((size_t)B_ * H_ * SLICES * NM * sizeof(bf16));
  bf16*  Q   = (bf16*)alloc((size_t)B_ * H_ * SLICES * NM * sizeof(bf16));
  float* OUT = (float*)alloc((size_t)B_ * H_ * SLICES * NM * sizeof(float));
  float* H1F = (float*)alloc((size_t)ROWS * C_ * sizeof(float));
  bf16*  H1B = (bf16*)alloc((size_t)ROWS * C_ * sizeof(bf16));

  k1_proj<<<ROWS / 16 / 4, 128, 0, stream>>>(h, wa1, wa2, wv1, wv2, A1, A2, V1R, V2R);
  k2_max<<<(2 * B_ * N_ * H_) / 256, 256, 0, stream>>>(A1, A2, M1, M2);
  k3_pack<<<B_ * H_ * 16 * 16, 256, 0, stream>>>(A1, A2, M1, M2, mask, V1R, V2R, P, Q);
  k4_tri<<<B_ * H_ * SLICES * 16, 256, 0, stream>>>(P, Q, OUT);
  k5_wo_ln1<<<B_ * N_ * (N_ / 16), 128, 0, stream>>>(OUT, h, wo, ln1s, ln1b, H1F, H1B);
  k6_mlp<<<ROWS / 16, 128, 0, stream>>>(H1F, H1B, w1, b1, w2, b2, ln2s, ln2b, out);
}
